// LLM_Decoder_1357209665611
// MI455X (gfx1250) — matrix operand order repaired and checked by reading, not compiled, not run on hardware
//
#include <hip/hip_runtime.h>
#include <hip/hip_bf16.h>
#include <math.h>

// ---------------------------------------------------------------------------
// Types for CDNA5 WMMA
// ---------------------------------------------------------------------------
typedef __attribute__((ext_vector_type(16))) __bf16 v16bf;
typedef __attribute__((ext_vector_type(8)))  float  v8f;

union BF16Frag { v16bf v; unsigned u[8]; };

#if defined(__has_builtin)
#if __has_builtin(__builtin_amdgcn_cvt_pk_bf16_f32)
#define HAVE_CVT_PK 1
#else
#define HAVE_CVT_PK 0
#endif
#else
#define HAVE_CVT_PK 0
#endif

// round-to-nearest-even bf16 in the high 16 bits
__device__ __forceinline__ unsigned rne_bits(float f) {
  unsigned u = __float_as_uint(f);
  return u + 0x7FFFu + ((u >> 16) & 1u);
}

// pack two floats into one dword of bf16 {b:a}
__device__ __forceinline__ unsigned pack_bf16(float a, float b) {
#if HAVE_CVT_PK
  auto r = __builtin_amdgcn_cvt_pk_bf16_f32(a, b);
  union { decltype(r) v; unsigned u; } cv;
  cv.v = r;
  return cv.u;
#else
  // result = { rne(b)[31:16], rne(a)[31:16] } via one v_perm_b32
  return __builtin_amdgcn_perm(rne_bits(b), rne_bits(a), 0x07060302u);
#endif
}

__device__ __forceinline__ float bf2f(unsigned short h) {
  return __uint_as_float(((unsigned)h) << 16);
}

// ---------------------------------------------------------------------------
// Generic bf16-WMMA GEMM: out = A[M,K] @ W[K,N] (+bias | += existing)
// Row-mapped output: orow = (m/seg_len)*seg_stride + seg_off + m%seg_len
// M%128==0, N%128==0, K%32==0 for all uses in this model.
// 256 threads = 8 waves (4x2); wave tile 32x64 = 2x4 WMMA 16x16 tiles.
// ---------------------------------------------------------------------------
#define BM 128
#define BN 128
#define BK 32
#define LDSA 34   // short stride: 68B = 17 dwords (odd) -> bank-conflict friendly

__global__ __launch_bounds__(256)
void gemm_bf16_wmma(const float* __restrict__ A, const float* __restrict__ W,
                    const float* __restrict__ bias, float* __restrict__ out,
                    int M, int N, int K, int op,
                    int seg_len, int seg_stride, int seg_off)
{
  __shared__ unsigned short As[BM * LDSA];
  __shared__ unsigned short Bs[BN * LDSA];
  const int tid  = threadIdx.x;
  const int lane = tid & 31;
  const int wave = tid >> 5;
  const int wm   = wave >> 1;     // 0..3 -> 32-row band
  const int wn   = wave & 1;      // 0..1 -> 64-col band
  const int hf   = lane >> 4;     // lane half
  const int l16  = lane & 15;
  const int m0 = blockIdx.y * BM;
  const int n0 = blockIdx.x * BN;
  (void)M;

  v8f acc[2][4];
#pragma unroll
  for (int i = 0; i < 2; i++)
#pragma unroll
    for (int j = 0; j < 4; j++)
#pragma unroll
      for (int r = 0; r < 8; r++) acc[i][j][r] = 0.0f;

  const int arow = tid >> 1;
  const int acol = (tid & 1) << 4;

  for (int k0 = 0; k0 < K; k0 += BK) {
    // stage A tile (128x32) as bf16, row-major [m][k]; pairs packed per dword
    {
      const float* src = A + (size_t)(m0 + arow) * K + (k0 + acol);
      unsigned short* dst = &As[arow * LDSA + acol];
#pragma unroll
      for (int e = 0; e < 16; e += 2)
        *(unsigned*)(&dst[e]) = pack_bf16(src[e], src[e + 1]);
    }
    // stage B tile (32x128) transposed: Bs[n][k]. Each thread reads float2
    // from two adjacent k rows and emits one packed dword per n (k-pair
    // contiguous in LDS) -> 1 pack + 1 ds_store_b32 per 2 elements.
    {
#pragma unroll
      for (int e = 0; e < 4; e++) {
        int lin = tid + (e << 8);            // 0..1023
        int kr  = lin >> 6;                  // k-pair index 0..15
        int np  = (lin & 63) << 1;           // n (even) 0..126
        const float2 r0 = *(const float2*)&W[(size_t)(k0 + 2 * kr)     * N + (n0 + np)];
        const float2 r1 = *(const float2*)&W[(size_t)(k0 + 2 * kr + 1) * N + (n0 + np)];
        *(unsigned*)&Bs[(np)     * LDSA + 2 * kr] = pack_bf16(r0.x, r1.x);
        *(unsigned*)&Bs[(np + 1) * LDSA + 2 * kr] = pack_bf16(r0.y, r1.y);
      }
    }
    if (k0 + BK < K) {
      __builtin_prefetch(A + (size_t)(m0 + arow) * K + (k0 + BK + acol), 0, 0);
      __builtin_prefetch(W + (size_t)(k0 + BK + (tid >> 7)) * N + n0 + ((tid & 1) << 6), 0, 0);
    }
    __syncthreads();

    // A frag: 16x32 bf16. lane half hf: pairs 0..3 -> K = hf*8 + 2p,
    // pairs 4..7 -> K = 16 + hf*8 + 2p  (ISA 16-bit A layout)
    BF16Frag af[2], bfr[4];
#pragma unroll
    for (int mt = 0; mt < 2; mt++) {
      const unsigned short* p = &As[(wm * 32 + mt * 16 + l16) * LDSA + (hf << 3)];
#pragma unroll
      for (int pr = 0; pr < 4; pr++) {
        af[mt].u[pr]     = *(const unsigned*)(&p[pr << 1]);
        af[mt].u[pr + 4] = *(const unsigned*)(&p[16 + (pr << 1)]);
      }
    }
    // B frag: 32x16 bf16. lane half hf: pairs 0..3 -> K = hf*8 + 2p,
    // pairs 4..7 -> K = 16 + hf*8 + 2p (same as A)
#pragma unroll
    for (int nt = 0; nt < 4; nt++) {
      const unsigned short* p = &Bs[(wn * 64 + nt * 16 + l16) * LDSA + (hf << 3)];
#pragma unroll
      for (int pr = 0; pr < 4; pr++) {
        bfr[nt].u[pr]     = *(const unsigned*)(&p[pr << 1]);
        bfr[nt].u[pr + 4] = *(const unsigned*)(&p[16 + (pr << 1)]);
      }
    }
#pragma unroll
    for (int mt = 0; mt < 2; mt++)
#pragma unroll
      for (int nt = 0; nt < 4; nt++)
        acc[mt][nt] = __builtin_amdgcn_wmma_f32_16x16x32_bf16(
            false, af[mt].v, false, bfr[nt].v, (short)0, acc[mt][nt], false, false);
    __syncthreads();
  }

  // epilogue: D 16x16 f32 layout -> row = r + 8*hf, col = l16
#pragma unroll
  for (int mt = 0; mt < 2; mt++) {
#pragma unroll
    for (int nt = 0; nt < 4; nt++) {
      int gn = n0 + wn * 64 + nt * 16 + l16;
      float bsv = (op == 0 && bias) ? bias[gn] : 0.0f;
#pragma unroll
      for (int r = 0; r < 8; r++) {
        int m = m0 + wm * 32 + mt * 16 + hf * 8 + r;
        int orow = (m / seg_len) * seg_stride + seg_off + (m % seg_len);
        size_t oidx = (size_t)orow * N + gn;
        float val = acc[mt][nt][r];
        if (op == 1) out[oidx] += val;
        else         out[oidx] = val + bsv;
      }
    }
  }
}

// ---------------------------------------------------------------------------
// prompt embedding gather: h[b*256+s] = embed_table[prompt_ids[b,s]]
// ---------------------------------------------------------------------------
__global__ __launch_bounds__(256)
void embed_gather(const int* __restrict__ ids, const float* __restrict__ table,
                  float* __restrict__ h)
{
  int s = blockIdx.x;   // 0..127
  int b = blockIdx.y;   // 0..7
  int id = ids[b * 128 + s];
  const float* src = table + (size_t)id * 4096;
  float* dst = h + ((size_t)(b * 256 + s)) * 4096;
  for (int d = threadIdx.x; d < 4096; d += 256) dst[d] = src[d];
}

// ---------------------------------------------------------------------------
// RMSNorm over 4096, one block per row
// ---------------------------------------------------------------------------
__global__ __launch_bounds__(256)
void rmsnorm_kernel(const float* __restrict__ in, const float* __restrict__ w,
                    float* __restrict__ out)
{
  __shared__ float red[256];
  int row = blockIdx.x;
  const float* x = in + (size_t)row * 4096;
  float ss = 0.f;
  for (int d = threadIdx.x; d < 4096; d += 256) { float v = x[d]; ss += v * v; }
  red[threadIdx.x] = ss;
  __syncthreads();
  for (int st = 128; st > 0; st >>= 1) {
    if (threadIdx.x < st) red[threadIdx.x] += red[threadIdx.x + st];
    __syncthreads();
  }
  float scale = rsqrtf(red[0] * (1.0f / 4096.0f) + 1e-5f);
  for (int d = threadIdx.x; d < 4096; d += 256)
    out[(size_t)row * 4096 + d] = x[d] * scale * w[d];
}

// ---------------------------------------------------------------------------
// RoPE in place, HF layout, HEAD_DIM=128, theta=10000. One block per token row.
// ---------------------------------------------------------------------------
__global__ __launch_bounds__(256)
void rope_kernel(float* __restrict__ x)
{
  int row = blockIdx.x;      // 0..2047
  int s   = row & 255;
  float* p = x + (size_t)row * 4096;
  const float kexp = -9.210340371976184f / 64.0f;   // -ln(10000)/64
  for (int i = threadIdx.x; i < 2048; i += 256) {   // 32 heads * 64 pairs
    int hh = i >> 6, j = i & 63;
    float inv = __expf(kexp * (float)j);
    float ang = (float)s * inv;
    float c = __cosf(ang), sn = __sinf(ang);
    float* q = p + hh * 128;
    float a = q[j], b2 = q[j + 64];
    q[j]      = a * c  - b2 * sn;
    q[j + 64] = b2 * c + a  * sn;
  }
}

// ---------------------------------------------------------------------------
// Attention: one block per (head, batch). K/V staged in LDS as bf16
// (stride 130 shorts -> conflict free). 8 waves; each wave owns q rows
// wave, wave+8, ... Softmax via wave32 shfl reductions.
// dyn LDS: 2*256*130*2 + 8*256*4 + 8*128*4 = 145408 bytes
// ---------------------------------------------------------------------------
#define KVLD 130
__global__ __launch_bounds__(256)
void attn_kernel(const float* __restrict__ q, const float* __restrict__ k,
                 const float* __restrict__ v, float* __restrict__ o)
{
  extern __shared__ char smem[];
  unsigned short* Ks = (unsigned short*)smem;
  unsigned short* Vs = Ks + 256 * KVLD;
  float* sc = (float*)(Vs + 256 * KVLD);
  float* qs = sc + 8 * 256;
  const int hd = blockIdx.x;   // head 0..31
  const int b  = blockIdx.y;   // batch 0..7
  const int tid = threadIdx.x, lane = tid & 31, wave = tid >> 5;
  const size_t base = ((size_t)(b * 256)) * 4096 + (size_t)hd * 128;

  // paired staging -> dword LDS stores (KVLD even, d even => 4B aligned)
  for (int i = tid * 2; i < 256 * 128; i += 512) {
    int row = i >> 7, d = i & 127;
    size_t g = base + (size_t)row * 4096 + d;
    *(unsigned*)&Ks[row * KVLD + d] = pack_bf16(k[g], k[g + 1]);
    *(unsigned*)&Vs[row * KVLD + d] = pack_bf16(v[g], v[g + 1]);
  }
  __syncthreads();

  float* mysc = sc + wave * 256;
  float* myq  = qs + wave * 128;
  const float scale = 0.088388347648318447f;   // 1/sqrt(128)

  for (int qi = wave; qi < 256; qi += 8) {
    for (int d = lane; d < 128; d += 32)
      myq[d] = q[base + (size_t)qi * 4096 + d];
    // intra-wave LDS RAW: DS ops are in-order per wave; compiler waits on use
    float p[8];
    float mloc = -3.0e38f;
#pragma unroll
    for (int j = 0; j < 8; j++) {
      int key = (j << 5) + lane;
      float dot = -3.0e38f;
      if (key <= qi) {
        float a = 0.f;
        for (int d = 0; d < 128; d++)
          a += bf2f(Ks[key * KVLD + d]) * myq[d];
        dot = a * scale;
      }
      p[j] = dot;
      mloc = fmaxf(mloc, dot);
    }
    for (int off = 16; off > 0; off >>= 1)
      mloc = fmaxf(mloc, __shfl_xor(mloc, off, 32));
    float ssum = 0.f;
#pragma unroll
    for (int j = 0; j < 8; j++) { p[j] = __expf(p[j] - mloc); ssum += p[j]; }
    for (int off = 16; off > 0; off >>= 1)
      ssum += __shfl_xor(ssum, off, 32);
    float inv = 1.0f / ssum;
#pragma unroll
    for (int j = 0; j < 8; j++) mysc[(j << 5) + lane] = p[j] * inv;

#pragma unroll
    for (int jj = 0; jj < 4; jj++) {
      int d = (jj << 5) + lane;
      float a = 0.f;
      for (int key = 0; key <= qi; key++)
        a += mysc[key] * bf2f(Vs[key * KVLD + d]);
      o[base + (size_t)qi * 4096 + d] = a;
    }
  }
}

// ---------------------------------------------------------------------------
// g = silu(g) * u, elementwise
// ---------------------------------------------------------------------------
__global__ void silu_mul_kernel(float* __restrict__ g, const float* __restrict__ u,
                                size_t n)
{
  size_t i = (size_t)blockIdx.x * blockDim.x + threadIdx.x;
  size_t stride = (size_t)gridDim.x * blockDim.x;
  for (; i < n; i += stride) {
    float x = g[i];
    g[i] = (x / (1.0f + __expf(-x))) * u[i];
  }
}

// ---------------------------------------------------------------------------
// zero fill
// ---------------------------------------------------------------------------
__global__ void zero_kernel(float* __restrict__ p, size_t n)
{
  size_t i = (size_t)blockIdx.x * blockDim.x + threadIdx.x;
  size_t stride = (size_t)gridDim.x * blockDim.x;
  for (; i < n; i += stride) p[i] = 0.0f;
}

// ---------------------------------------------------------------------------
// mean pool over S=256 into rows 0..7 of pooled_pad[128,4096]
// ---------------------------------------------------------------------------
__global__ __launch_bounds__(256)
void pool_kernel(const float* __restrict__ x, float* __restrict__ pooled)
{
  int b = blockIdx.x >> 4;
  int d = ((blockIdx.x & 15) << 8) + threadIdx.x;
  float s = 0.f;
  for (int t = 0; t < 256; t++)
    s += x[((size_t)(b * 256 + t)) * 4096 + d];
  pooled[(size_t)b * 4096 + d] = s * (1.0f / 256.0f);
}

// ---------------------------------------------------------------------------
// out[b] = y1[b,:] . out2_W[:,0] + out2_b[0]   (8 blocks)
// ---------------------------------------------------------------------------
__global__ __launch_bounds__(256)
void head2_kernel(const float* __restrict__ y1, const float* __restrict__ W2,
                  const float* __restrict__ b2, float* __restrict__ out)
{
  __shared__ float red[256];
  int b = blockIdx.x;
  float s = 0.f;
  for (int j = threadIdx.x; j < 768; j += 256)
    s += y1[b * 768 + j] * W2[j];
  red[threadIdx.x] = s;
  __syncthreads();
  for (int st = 128; st > 0; st >>= 1) {
    if (threadIdx.x < st) red[threadIdx.x] += red[threadIdx.x + st];
    __syncthreads();
  }
  if (threadIdx.x == 0) out[b] = red[0] + b2[0];
}

// ---------------------------------------------------------------------------
static inline void gemm(const float* A, const float* W, const float* bias,
                        float* out, int M, int N, int K, int op,
                        int seg_len, int seg_stride, int seg_off,
                        hipStream_t stream)
{
  dim3 grid(N / BN, M / BM);
  gemm_bf16_wmma<<<grid, 256, 0, stream>>>(A, W, bias, out, M, N, K, op,
                                           seg_len, seg_stride, seg_off);
}

extern "C" void kernel_launch(void* const* d_in, const int* in_sizes, int n_in,
                              void* d_out, int out_size, void* d_ws, size_t ws_size,
                              hipStream_t stream)
{
  (void)in_sizes; (void)n_in; (void)out_size; (void)ws_size;
  const float* text_embed   = (const float*)d_in[0];
  const float* vision_embed = (const float*)d_in[1];
  const int*   prompt_ids   = (const int*)d_in[2];
  const float* in_W  = (const float*)d_in[3];
  const float* in_b  = (const float*)d_in[4];
  const float* table = (const float*)d_in[5];
  const float* Wq = (const float*)d_in[6];
  const float* Wk = (const float*)d_in[7];
  const float* Wv = (const float*)d_in[8];
  const float* Wo = (const float*)d_in[9];
  const float* ln1 = (const float*)d_in[10];
  const float* ln2 = (const float*)d_in[11];
  const float* Wg = (const float*)d_in[12];
  const float* Wu = (const float*)d_in[13];
  const float* Wd = (const float*)d_in[14];
  const float* fnw = (const float*)d_in[15];
  const float* o1W = (const float*)d_in[16];
  const float* o1b = (const float*)d_in[17];
  const float* o2W = (const float*)d_in[18];
  const float* o2b = (const float*)d_in[19];
  float* out = (float*)d_out;

  float* ws = (float*)d_ws;
  const size_t HSZ = (size_t)2048 * 4096;       // [B*S, D_LLM]
  const size_t FSZ = (size_t)2048 * 11008;      // [B*S, D_FF]
  float* h      = ws;
  float* xb     = h  + HSZ;
  float* qb     = xb + HSZ;
  float* kb     = qb + HSZ;
  float* vb     = kb + HSZ;
  float* ob     = vb + HSZ;
  float* gb     = ob + HSZ;
  float* ub     = gb + FSZ;
  float* pooled = ub + FSZ;                     // padded to [128, 4096]
  float* y1     = pooled + (size_t)128 * 4096;  // [128, 768] (rows 0..7 used)

  // h = concat(embed_table[ids], te@in_W+b, ve@in_W+b) per batch
  embed_gather<<<dim3(128, 8), 256, 0, stream>>>(prompt_ids, table, h);
  gemm(text_embed,   in_W, in_b, h, 512, 4096, 768, 0, 64, 256, 128, stream);
  gemm(vision_embed, in_W, in_b, h, 512, 4096, 768, 0, 64, 256, 192, stream);

  const size_t WSQ = (size_t)4096 * 4096;
  const size_t WSF = (size_t)4096 * 11008;
  const int attn_smem = 2 * 256 * KVLD * 2 + 8 * 256 * 4 + 8 * 128 * 4;

  for (int l = 0; l < 2; l++) {
    rmsnorm_kernel<<<2048, 256, 0, stream>>>(h, ln1 + l * 4096, xb);
    gemm(xb, Wq + l * WSQ, nullptr, qb, 2048, 4096, 4096, 0, 2048, 0, 0, stream);
    gemm(xb, Wk + l * WSQ, nullptr, kb, 2048, 4096, 4096, 0, 2048, 0, 0, stream);
    gemm(xb, Wv + l * WSQ, nullptr, vb, 2048, 4096, 4096, 0, 2048, 0, 0, stream);
    rope_kernel<<<2048, 256, 0, stream>>>(qb);
    rope_kernel<<<2048, 256, 0, stream>>>(kb);
    attn_kernel<<<dim3(32, 8), 256, attn_smem, stream>>>(qb, kb, vb, ob);
    gemm(ob, Wo + l * WSQ, nullptr, h, 2048, 4096, 4096, 1, 2048, 0, 0, stream);
    rmsnorm_kernel<<<2048, 256, 0, stream>>>(h, ln2 + l * 4096, xb);
    gemm(xb, Wg + l * WSF, nullptr, gb, 2048, 11008, 4096, 0, 2048, 0, 0, stream);
    gemm(xb, Wu + l * WSF, nullptr, ub, 2048, 11008, 4096, 0, 2048, 0, 0, stream);
    silu_mul_kernel<<<8192, 256, 0, stream>>>(gb, ub, FSZ);
    gemm(gb, Wd + l * WSF, nullptr, h, 2048, 4096, 11008, 1, 2048, 0, 0, stream);
  }

  rmsnorm_kernel<<<2048, 256, 0, stream>>>(h, fnw, xb);
  zero_kernel<<<1024, 256, 0, stream>>>(pooled, (size_t)128 * 4096);
  pool_kernel<<<128, 256, 0, stream>>>(xb, pooled);
  // y1[0..7,:] = pooled @ out1_W + out1_b   (WMMA GEMM on padded M=128)
  gemm(pooled, o1W, o1b, y1, 128, 768, 4096, 0, 128, 0, 0, stream);
  head2_kernel<<<8, 256, 0, stream>>>(y1, o2W, o2b, out);
}
